// CausalAttentionKVCache_73083163508874
// MI455X (gfx1250) — matrix operand order repaired and checked by reading, not compiled, not run on hardware
//
#include <hip/hip_runtime.h>
#include <hip/hip_bf16.h>

typedef __attribute__((ext_vector_type(16))) _Float16 v16h;
typedef __attribute__((ext_vector_type(8)))  _Float16 v8h;
typedef __attribute__((ext_vector_type(4)))  _Float16 v4h;
typedef __attribute__((ext_vector_type(8)))  float    v8f;
typedef __attribute__((ext_vector_type(4)))  float    v4f;

#define HIDDEN   1024
#define THREE_H  3072
#define NH       16
#define HS       64
#define BATCH    2
#define SEQ      2048
#define MROWS    (BATCH*SEQ)          // 4096
#define BTD      (BATCH*SEQ*HIDDEN)   // 4194304 = 2^22
#define HALF_BUF ((size_t)BATCH*NH*SEQ*HS)      // 4194304 halfs
#define XH_ELEMS ((size_t)MROWS*HIDDEN)         // 4194304 halfs
#define WT_ELEMS ((size_t)THREE_H*HIDDEN)       // 3145728 halfs

// ---------------------------------------------------------------------------
// Prep 1: x (fp32) -> xh (f16), same [M][K] layout. Coalesced float4 loads.
// ---------------------------------------------------------------------------
__global__ __launch_bounds__(256)
void cvt_x_kernel(const float* __restrict__ x, _Float16* __restrict__ xh)
{
    const size_t i = ((size_t)blockIdx.x * 256 + threadIdx.x) * 4;
    v4f f = *(const v4f*)(x + i);
    v4h h;
#pragma unroll
    for (int j = 0; j < 4; ++j) h[j] = (_Float16)f[j];
    *(v4h*)(xh + i) = h;
}

// ---------------------------------------------------------------------------
// Prep 2: W [K=1024][N=3072] fp32 -> Wt [N][K] f16 via LDS-tiled transpose.
// Block = 256 threads handles a 32(k) x 32(n) tile; both sides coalesced.
// ---------------------------------------------------------------------------
__global__ __launch_bounds__(256)
void cvt_w_kernel(const float* __restrict__ W, _Float16* __restrict__ Wt)
{
    __shared__ float tile[32][33];
    const int tx = threadIdx.x & 31;   // fast dim
    const int ty = threadIdx.x >> 5;   // 0..7
    const int ntile = blockIdx.x;      // 0..95  (3072/32)
    const int ktile = blockIdx.y;      // 0..31  (1024/32)

#pragma unroll
    for (int r = 0; r < 4; ++r) {
        const int kk = ty + r * 8;
        tile[kk][tx] = W[(size_t)(ktile * 32 + kk) * THREE_H + ntile * 32 + tx];
    }
    __syncthreads();
#pragma unroll
    for (int r = 0; r < 4; ++r) {
        const int nn = ty + r * 8;
        Wt[(size_t)(ntile * 32 + nn) * HIDDEN + ktile * 32 + tx] =
            (_Float16)tile[tx][nn];
    }
}

// ---------------------------------------------------------------------------
// Kernel 1: QKV GEMM  y = x @ W + b  (M=4096, N=3072, K=1024).
// All-f16 inputs, fp32 accumulate. Each wave computes a 16x64 strip:
// A fragment loaded once per k-step, reused across 4 WMMAs; every fragment
// load is contiguous (A: 2x16B, B: 4x32B per k-step). Epilogue reproduces
// the reference's reshape(3,B,T,hs,nh).transpose(0,1,4,2,3) by decomposing
// each output element's flat index, scattering f16 into packed Qh/Kh
// ([bh][t][hs]) and V-transposed Vt ([bh][hs][t]).
// ---------------------------------------------------------------------------
__global__ __launch_bounds__(256)
void qkv_gemm_kernel(const _Float16* __restrict__ xh, const _Float16* __restrict__ Wt,
                     const float* __restrict__ bias,
                     _Float16* __restrict__ Qh, _Float16* __restrict__ Kh,
                     _Float16* __restrict__ Vt)
{
    const int lane  = threadIdx.x & 31;
    const int wid   = threadIdx.x >> 5;
    const int ngrp  = blockIdx.x;              // 0..47  (64 cols each)
    const int mtile = blockIdx.y * 8 + wid;    // 0..255
    const bool hi   = lane >= 16;
    const int  ln   = lane & 15;
    const int  klo  = hi ? 16 : 0;             // A-frag per-lane K offset
    const int  kbo  = hi ? 16 : 0;             // B-frag per-lane K offset

    const _Float16* arow = xh + (size_t)(mtile * 16 + ln) * HIDDEN;
    const _Float16* b0 = Wt + (size_t)(ngrp * 64 +      ln) * HIDDEN + kbo;
    const _Float16* b1 = Wt + (size_t)(ngrp * 64 + 16 + ln) * HIDDEN + kbo;
    const _Float16* b2 = Wt + (size_t)(ngrp * 64 + 32 + ln) * HIDDEN + kbo;
    const _Float16* b3 = Wt + (size_t)(ngrp * 64 + 48 + ln) * HIDDEN + kbo;

    v8f acc0 = {}, acc1 = {}, acc2 = {}, acc3 = {};

    for (int k0 = 0; k0 < HIDDEN; k0 += 32) {
        __builtin_prefetch(arow + k0 + 128, 0, 0);
        // A fragment: per-lane K runs klo..klo+15 contiguous
        v8h l0 = *(const v8h*)(arow + k0 + klo);
        v8h l1 = *(const v8h*)(arow + k0 + 8 + klo);
        v16h a;
#pragma unroll
        for (int j = 0; j < 8; ++j) { a[j] = l0[j]; a[8 + j] = l1[j]; }
        // B fragments: lane holds column n, K = kbo..kbo+15 contiguous
        v16h bf0 = *(const v16h*)(b0 + k0);
        v16h bf1 = *(const v16h*)(b1 + k0);
        v16h bf2 = *(const v16h*)(b2 + k0);
        v16h bf3 = *(const v16h*)(b3 + k0);
        acc0 = __builtin_amdgcn_wmma_f32_16x16x32_f16(false, a, false, bf0, (short)0, acc0, false, false);
        acc1 = __builtin_amdgcn_wmma_f32_16x16x32_f16(false, a, false, bf1, (short)0, acc1, false, false);
        acc2 = __builtin_amdgcn_wmma_f32_16x16x32_f16(false, a, false, bf2, (short)0, acc2, false, false);
        acc3 = __builtin_amdgcn_wmma_f32_16x16x32_f16(false, a, false, bf3, (short)0, acc3, false, false);
    }

    // ---- epilogue: bias + exact reference reshape scatter (f16)
#pragma unroll
    for (int c = 0; c < 4; ++c) {
        const v8f acc = (c == 0) ? acc0 : (c == 1) ? acc1 : (c == 2) ? acc2 : acc3;
        const int n = ngrp * 64 + c * 16 + ln;
        const float bn = bias[n];
#pragma unroll
        for (int r = 0; r < 8; ++r) {
            const int mrow = mtile * 16 + r + (hi ? 8 : 0); // C layout: M=r (+8 hi)
            const int f    = mrow * THREE_H + n;            // flat index into y
            const int s    = f >> 22;                       // /BTD -> 0:q 1:k 2:v
            const int rem  = f & (BTD - 1);
            const int bb   = rem >> 21;                     // /TD
            const int t2   = (rem >> 10) & (SEQ - 1);
            const int r3   = rem & (HIDDEN - 1);
            const int h    = r3 >> 4;
            const int hd   = r3 & 15;
            const int bh   = bb * NH + hd;
            const _Float16 v = (_Float16)(acc[r] + bn);
            if (s == 0)      Qh[((size_t)bh * SEQ + t2) * HS + h] = v;
            else if (s == 1) Kh[((size_t)bh * SEQ + t2) * HS + h] = v;
            else             Vt[((size_t)bh * HS + h) * SEQ + t2] = v;
        }
    }
}

// ---------------------------------------------------------------------------
// Kernel 2: flash-style causal attention. One wave32 per (head, 16-query
// tile); 1-wave workgroups so __syncthreads() is wave-local. Per 32-key
// block: S = Q*K^T via 4 WMMAs, online softmax (shfl_xor row reductions over
// the 16-lane C-layout groups), P transposed C->A layout through LDS, then
// O += P*V via 4 WMMAs against the pre-transposed V.
// ---------------------------------------------------------------------------
__global__ __launch_bounds__(32)
void attn_kernel(const _Float16* __restrict__ Qh, const _Float16* __restrict__ Kh,
                 const _Float16* __restrict__ Vt, float* __restrict__ out)
{
    __shared__ _Float16 Plds[16 * 32];

    const int  lane  = threadIdx.x;
    const int  qtile = blockIdx.x;       // 0..127
    const int  bh    = blockIdx.y;       // 0..31  (= b*16 + head)
    const int  b2    = bh >> 4;
    const int  hd    = bh & 15;
    const bool hiw   = lane >= 16;
    const int  ln    = lane & 15;
    const int  qbase = qtile * 16;

    // ---- load Q A-fragments once (hs=64 -> two K=32 fragments)
    const _Float16* qrow = Qh + ((size_t)bh * SEQ + qbase + ln) * HS;
    v16h aq0, aq1;
    {
        const int off = hiw ? 16 : 0;
        v8h l0 = *(const v8h*)(qrow + off);
        v8h l1 = *(const v8h*)(qrow + 8 + off);
        v8h l2 = *(const v8h*)(qrow + 32 + off);
        v8h l3 = *(const v8h*)(qrow + 40 + off);
#pragma unroll
        for (int j = 0; j < 8; ++j) {
            aq0[j] = l0[j]; aq0[8 + j] = l1[j];
            aq1[j] = l2[j]; aq1[8 + j] = l3[j];
        }
    }

    const _Float16* kb = Kh + (size_t)bh * SEQ * HS;
    const _Float16* vb = Vt + (size_t)bh * HS * SEQ;

    v8f o0 = {}, o1 = {}, o2 = {}, o3 = {};
    float mrow[8], lrow[8];
#pragma unroll
    for (int r = 0; r < 8; ++r) { mrow[r] = -__builtin_inff(); lrow[r] = 0.f; }

    const float scale   = 0.125f;                 // 1/sqrt(64)
    const int   nblocks = (qbase + 16 + 31) >> 5; // causal: keys 0..qbase+15

    for (int blk = 0; blk < nblocks; ++blk) {
        const int s0 = blk * 32;

        // ---- S tiles (16x16 each) over 32 keys
        v8f S0 = {}, S1 = {};
        {
            const _Float16* kr0 = kb + (size_t)(s0 + ln) * HS + (hiw ? 16 : 0);
            const _Float16* kr1 = kb + (size_t)(s0 + 16 + ln) * HS + (hiw ? 16 : 0);
            v16h bk00 = *(const v16h*)(kr0);
            v16h bk01 = *(const v16h*)(kr0 + 32);
            v16h bk10 = *(const v16h*)(kr1);
            v16h bk11 = *(const v16h*)(kr1 + 32);
            S0 = __builtin_amdgcn_wmma_f32_16x16x32_f16(false, aq0, false, bk00, (short)0, S0, false, false);
            S0 = __builtin_amdgcn_wmma_f32_16x16x32_f16(false, aq1, false, bk01, (short)0, S0, false, false);
            S1 = __builtin_amdgcn_wmma_f32_16x16x32_f16(false, aq0, false, bk10, (short)0, S1, false, false);
            S1 = __builtin_amdgcn_wmma_f32_16x16x32_f16(false, aq1, false, bk11, (short)0, S1, false, false);
        }

        // ---- causal mask + scale + online softmax (row stats per C row)
        float p0[8], p1[8];
#pragma unroll
        for (int r = 0; r < 8; ++r) {
            const int t = qbase + r + (hiw ? 8 : 0);
            float v0 = S0[r] * scale;
            float v1 = S1[r] * scale;
            if (s0 + ln > t)      v0 = -__builtin_inff();
            if (s0 + 16 + ln > t) v1 = -__builtin_inff();
            float bm = fmaxf(v0, v1);
#pragma unroll
            for (int off = 1; off < 16; off <<= 1)
                bm = fmaxf(bm, __shfl_xor(bm, off, 32));
            const float mnew  = fmaxf(mrow[r], bm);
            const float alpha = __expf(mrow[r] - mnew);
            mrow[r] = mnew;
            const float e0 = __expf(v0 - mnew);
            const float e1 = __expf(v1 - mnew);
            float rs = e0 + e1;
#pragma unroll
            for (int off = 1; off < 16; off <<= 1)
                rs += __shfl_xor(rs, off, 32);
            lrow[r] = lrow[r] * alpha + rs;
            o0[r] *= alpha; o1[r] *= alpha; o2[r] *= alpha; o3[r] *= alpha;
            p0[r] = e0; p1[r] = e1;
        }

        // ---- transpose P from C layout to A layout via LDS
#pragma unroll
        for (int r = 0; r < 8; ++r) {
            const int mr = r + (hiw ? 8 : 0);
            Plds[mr * 32 + ln]      = (_Float16)p0[r];
            Plds[mr * 32 + 16 + ln] = (_Float16)p1[r];
        }
        __syncthreads();
        v16h ap;
        {
            const int off = hiw ? 16 : 0;
            const _Float16* pr = &Plds[ln * 32];
            v8h l0 = *(const v8h*)(pr + off);
            v8h l1 = *(const v8h*)(pr + 8 + off);
#pragma unroll
            for (int j = 0; j < 8; ++j) { ap[j] = l0[j]; ap[8 + j] = l1[j]; }
        }
        __syncthreads();

        // ---- O += P(16x32) * V(32x64), V pre-transposed -> contiguous loads
        {
            const _Float16* vp = vb + s0 + (hiw ? 16 : 0);
            v16h bv0 = *(const v16h*)(vp + (size_t)(ln) * SEQ);
            v16h bv1 = *(const v16h*)(vp + (size_t)(16 + ln) * SEQ);
            v16h bv2 = *(const v16h*)(vp + (size_t)(32 + ln) * SEQ);
            v16h bv3 = *(const v16h*)(vp + (size_t)(48 + ln) * SEQ);
            o0 = __builtin_amdgcn_wmma_f32_16x16x32_f16(false, ap, false, bv0, (short)0, o0, false, false);
            o1 = __builtin_amdgcn_wmma_f32_16x16x32_f16(false, ap, false, bv1, (short)0, o1, false, false);
            o2 = __builtin_amdgcn_wmma_f32_16x16x32_f16(false, ap, false, bv2, (short)0, o2, false, false);
            o3 = __builtin_amdgcn_wmma_f32_16x16x32_f16(false, ap, false, bv3, (short)0, o3, false, false);
        }
    }

    // ---- normalize and write out[b, t, hd*64 + h]  (fp32)
#pragma unroll
    for (int r = 0; r < 8; ++r) {
        const float rinv = 1.0f / lrow[r];
        const int t = qbase + r + (hiw ? 8 : 0);
        float* op = out + ((size_t)(b2 * SEQ + t)) * HIDDEN + hd * HS;
        op[ln]      = o0[r] * rinv;
        op[16 + ln] = o1[r] * rinv;
        op[32 + ln] = o2[r] * rinv;
        op[48 + ln] = o3[r] * rinv;
    }
}

// ---------------------------------------------------------------------------
extern "C" void kernel_launch(void* const* d_in, const int* in_sizes, int n_in,
                              void* d_out, int out_size, void* d_ws, size_t ws_size,
                              hipStream_t stream) {
    const float* x    = (const float*)d_in[0];   // [B, T, 1024]
    const float* Wqkv = (const float*)d_in[1];   // [1024, 3072]
    const float* bqkv = (const float*)d_in[2];   // [3072]
    float* out = (float*)d_out;                  // [B, T, 1024]

    _Float16* Qh = (_Float16*)d_ws;              // [32][2048][64]   8 MB
    _Float16* Kh = Qh + HALF_BUF;                // [32][2048][64]   8 MB
    _Float16* Vt = Kh + HALF_BUF;                // [32][64][2048]   8 MB
    _Float16* xh = Vt + HALF_BUF;                // [4096][1024]     8 MB
    _Float16* Wt = xh + XH_ELEMS;                // [3072][1024]     6 MB
    (void)in_sizes; (void)n_in; (void)out_size; (void)ws_size;

    cvt_x_kernel<<<dim3(XH_ELEMS / (256 * 4)), 256, 0, stream>>>(x, xh);
    cvt_w_kernel<<<dim3(96, 32), 256, 0, stream>>>(Wqkv, Wt);
    qkv_gemm_kernel<<<dim3(48, 32), 256, 0, stream>>>(xh, Wt, bqkv, Qh, Kh, Vt);
    attn_kernel<<<dim3(128, 32), 32, 0, stream>>>(Qh, Kh, Vt, out);
}
